// FixCapsNet_22033182228898
// MI455X (gfx1250) — compile-verified, matrix operand order checked by reading, not run on hardware
//
#include <hip/hip_runtime.h>
#include <hip/hip_bf16.h>

// Problem constants
#define C_CH   4096   // in_channels (i)
#define J_UN   10     // num_units  (j)
#define D_SZ   16     // unit_size  (d)
#define K_SZ   32     // in_units   (k)
#define B_SZ   128    // batch      (b)
#define JD     160    // J_UN * D_SZ

#define CH_PER_WG 16
#define THREADS   256
#define XS_CH_STRIDE 2052   // 16*128 dwords per channel + 4 pad (bank-conflict break)

typedef __bf16 v16bf __attribute__((ext_vector_type(16)));
typedef float  v8f   __attribute__((ext_vector_type(8)));

__device__ __forceinline__ unsigned pk2bf(float lo, float hi) {
    __hip_bfloat16 l = __float2bfloat16(lo);
    __hip_bfloat16 h = __float2bfloat16(hi);
    unsigned short ls, hs;
    __builtin_memcpy(&ls, &l, 2);
    __builtin_memcpy(&hs, &h, 2);
    return (unsigned)ls | ((unsigned)hs << 16);
}

// Build WMMA A operand (16x32 bf16) for one row m of W_{i,j}[16x32], fp32 -> bf16,
// with an optional f32 scale folded in before conversion.
// Lane layout (ISA 7.12.2, 16-bit A 16x32): lanes 0-15 hold K{0..7,16..23},
// lanes 16-31 hold K{8..15,24..31}, packed as (even,odd) bf16 pairs per dword.
__device__ __forceinline__ v16bf build_A(const float* __restrict__ Wrow, int k0,
                                         float scale) {
    const float4* p = (const float4*)Wrow;
    float4 f0 = p[(k0 >> 2) + 0];   // k0   .. k0+3
    float4 f1 = p[(k0 >> 2) + 1];   // k0+4 .. k0+7
    float4 f2 = p[(k0 >> 2) + 4];   // k0+16.. k0+19
    float4 f3 = p[(k0 >> 2) + 5];   // k0+20.. k0+23
    union { unsigned u[8]; v16bf v; } A;
    A.u[0] = pk2bf(scale * f0.x, scale * f0.y);
    A.u[1] = pk2bf(scale * f0.z, scale * f0.w);
    A.u[2] = pk2bf(scale * f1.x, scale * f1.y);
    A.u[3] = pk2bf(scale * f1.z, scale * f1.w);
    A.u[4] = pk2bf(scale * f2.x, scale * f2.y);
    A.u[5] = pk2bf(scale * f2.z, scale * f2.w);
    A.u[6] = pk2bf(scale * f3.x, scale * f3.y);
    A.u[7] = pk2bf(scale * f3.z, scale * f3.w);
    return A.v;
}

// Build WMMA B operand (32x16 bf16) from LDS-staged x-channel (packed bf16 pairs,
// layout xs_ch[kp*128 + b], kp = K/2). Symmetric lane layout to A.
__device__ __forceinline__ v16bf build_B(const unsigned* xs_ch, int ntile, int lane) {
    int bcol = ntile * 16 + (lane & 15);
    int kpb  = (lane < 16) ? 0 : 4;
    union { unsigned u[8]; v16bf v; } Bm;
#pragma unroll
    for (int r = 0; r < 4; ++r) Bm.u[r]     = xs_ch[(kpb + r)     * B_SZ + bcol];
#pragma unroll
    for (int r = 0; r < 4; ++r) Bm.u[4 + r] = xs_ch[(kpb + 8 + r) * B_SZ + bcol];
    return Bm.v;
}

// Cooperatively stage x[:, :, i0:i0+16] into LDS as packed bf16 pairs.
// Global loads coalesce across channels (i innermost in memory).
__device__ __forceinline__ void stage_x(const float* __restrict__ x, int i0,
                                        unsigned* xs, int tid) {
    for (int e = tid; e < CH_PER_WG * 16 * B_SZ; e += THREADS) {
        int il   = e & (CH_PER_WG - 1);
        int rest = e >> 4;
        int kp   = rest & 15;
        int b    = rest >> 4;
        float lo = x[(size_t)(b * K_SZ + 2 * kp    ) * C_CH + (i0 + il)];
        float hi = x[(size_t)(b * K_SZ + 2 * kp + 1) * C_CH + (i0 + il)];
        xs[il * XS_CH_STRIDE + kp * B_SZ + b] = pk2bf(lo, hi);
    }
}

// Pass A: s[j,d,b] += c[i,j] * u_hat[b,i,j,d], u_hat recomputed via bf16 WMMA.
// Each wave owns 2 channels; their contributions to the same s tile are chained
// through the WMMA C operand (native accumulate), halving ds_add traffic.
__global__ __launch_bounds__(THREADS)
void pass_s_kernel(const float* __restrict__ x, const float* __restrict__ W,
                   const float* __restrict__ c, float* __restrict__ s) {
    extern __shared__ unsigned smem[];
    unsigned* xs   = smem;                                      // 16 * 2052 dwords
    float* s_loc   = (float*)(smem + CH_PER_WG * XS_CH_STRIDE); // 160*128 floats
    int tid = threadIdx.x, lane = tid & 31, wave = tid >> 5;
    int i0 = blockIdx.x * CH_PER_WG;

    stage_x(x, i0, xs, tid);
    for (int e = tid; e < JD * B_SZ; e += THREADS) s_loc[e] = 0.f;
    __syncthreads();

    int k0 = (lane < 16) ? 0 : 8;
    int m  = lane & 15;
    int mbase = (lane < 16) ? 0 : 8;

    int il0 = wave * 2, il1 = wave * 2 + 1;
    int ch0 = i0 + il0, ch1 = i0 + il1;
    const unsigned* xs0 = xs + il0 * XS_CH_STRIDE;
    const unsigned* xs1 = xs + il1 * XS_CH_STRIDE;

    for (int j = 0; j < J_UN; ++j) {
        float c0 = c[ch0 * J_UN + j];
        float c1 = c[ch1 * J_UN + j];
        v16bf A0 = build_A(W + ((size_t)(ch0 * J_UN + j) * D_SZ + m) * K_SZ, k0, c0);
        v16bf A1 = build_A(W + ((size_t)(ch1 * J_UN + j) * D_SZ + m) * K_SZ, k0, c1);
        for (int nt = 0; nt < 8; ++nt) {
            v16bf B0 = build_B(xs0, nt, lane);
            v8f acc = {};
            acc = __builtin_amdgcn_wmma_f32_16x16x32_bf16(
                    false, A0, false, B0, (short)0, acc, false, false);
            v16bf B1 = build_B(xs1, nt, lane);
            acc = __builtin_amdgcn_wmma_f32_16x16x32_bf16(
                    false, A1, false, B1, (short)0, acc, false, false);
            int col = nt * 16 + (lane & 15);
#pragma unroll
            for (int r = 0; r < 8; ++r)
                unsafeAtomicAdd(&s_loc[(j * D_SZ + mbase + r) * B_SZ + col], acc[r]);
        }
    }
    __syncthreads();
    for (int e = tid; e < JD * B_SZ; e += THREADS)
        unsafeAtomicAdd(&s[e], s_loc[e]);
}

// Pass B: agree[i,j] = (1/B) * sum_{b,d} u_hat[b,i,j,d] * v[j,d,b] (u_hat recomputed).
// Both channels of a wave processed per (j,nt) so v_loc LDS reads are shared.
__global__ __launch_bounds__(THREADS)
void pass_agree_kernel(const float* __restrict__ x, const float* __restrict__ W,
                       const float* __restrict__ v, float* __restrict__ agree) {
    extern __shared__ unsigned smem[];
    unsigned* xs = smem;
    float* v_loc = (float*)(smem + CH_PER_WG * XS_CH_STRIDE); // 160*128 floats
    int tid = threadIdx.x, lane = tid & 31, wave = tid >> 5;
    int i0 = blockIdx.x * CH_PER_WG;

    stage_x(x, i0, xs, tid);
    for (int e = tid; e < JD * B_SZ; e += THREADS) v_loc[e] = v[e];
    __syncthreads();

    int k0 = (lane < 16) ? 0 : 8;
    int m  = lane & 15;
    int mbase = (lane < 16) ? 0 : 8;

    int il0 = wave * 2, il1 = wave * 2 + 1;
    int ch0 = i0 + il0, ch1 = i0 + il1;
    const unsigned* xs0 = xs + il0 * XS_CH_STRIDE;
    const unsigned* xs1 = xs + il1 * XS_CH_STRIDE;

    for (int j = 0; j < J_UN; ++j) {
        v16bf A0 = build_A(W + ((size_t)(ch0 * J_UN + j) * D_SZ + m) * K_SZ, k0, 1.0f);
        v16bf A1 = build_A(W + ((size_t)(ch1 * J_UN + j) * D_SZ + m) * K_SZ, k0, 1.0f);
        float p0 = 0.f, p1 = 0.f;
        for (int nt = 0; nt < 8; ++nt) {
            v16bf B0 = build_B(xs0, nt, lane);
            v8f d0 = {};
            d0 = __builtin_amdgcn_wmma_f32_16x16x32_bf16(
                    false, A0, false, B0, (short)0, d0, false, false);
            v16bf B1 = build_B(xs1, nt, lane);
            v8f d1 = {};
            d1 = __builtin_amdgcn_wmma_f32_16x16x32_bf16(
                    false, A1, false, B1, (short)0, d1, false, false);
            int col = nt * 16 + (lane & 15);
#pragma unroll
            for (int r = 0; r < 8; ++r) {
                float vv = v_loc[(j * D_SZ + mbase + r) * B_SZ + col];
                p0 += d0[r] * vv;
                p1 += d1[r] * vv;
            }
        }
#pragma unroll
        for (int off = 16; off > 0; off >>= 1) {
            p0 += __shfl_xor(p0, off, 32);
            p1 += __shfl_xor(p1, off, 32);
        }
        if (lane == 0) {
            agree[ch0 * J_UN + j] = p0 * (1.0f / (float)B_SZ);
            agree[ch1 * J_UN + j] = p1 * (1.0f / (float)B_SZ);
        }
    }
}

// b_ij += agree; c = softmax over channel axis (per j); zero s for next pass.
__global__ __launch_bounds__(320)
void compute_c_kernel(float* __restrict__ b_ij, const float* __restrict__ agree,
                      float* __restrict__ c, float* __restrict__ s) {
    int tid = threadIdx.x;
    for (int e = tid; e < C_CH * J_UN; e += 320) b_ij[e] += agree[e];
    for (int e = tid; e < JD * B_SZ; e += 320) s[e] = 0.f;
    __syncthreads();
    int wave = tid >> 5, lane = tid & 31;
    int j = wave;                       // 10 waves, one j each
    float mx = -1e30f;
    for (int i = lane; i < C_CH; i += 32) mx = fmaxf(mx, b_ij[i * J_UN + j]);
#pragma unroll
    for (int off = 16; off > 0; off >>= 1) mx = fmaxf(mx, __shfl_xor(mx, off, 32));
    float sum = 0.f;
    for (int i = lane; i < C_CH; i += 32) sum += expf(b_ij[i * J_UN + j] - mx);
#pragma unroll
    for (int off = 16; off > 0; off >>= 1) sum += __shfl_xor(sum, off, 32);
    float inv = 1.f / sum;
    for (int i = lane; i < C_CH; i += 32)
        c[i * J_UN + j] = expf(b_ij[i * J_UN + j] - mx) * inv;
}

// squash over class axis J; write v (layout [j][d][b] for pass_agree) and output.
__global__ __launch_bounds__(256)
void squash_kernel(const float* __restrict__ s, float* __restrict__ v,
                   float* __restrict__ out) {
    int tid = threadIdx.x;
    for (int p = tid; p < B_SZ * D_SZ; p += 256) {
        int b = p >> 4, d = p & 15;
        float msq = 0.f;
#pragma unroll
        for (int j = 0; j < J_UN; ++j) {
            float sv = s[(j * D_SZ + d) * B_SZ + b];
            msq += sv * sv;
        }
        float scale = msq / ((1.f + msq) * sqrtf(msq));
#pragma unroll
        for (int j = 0; j < J_UN; ++j) {
            float vv = scale * s[(j * D_SZ + d) * B_SZ + b];
            v[(j * D_SZ + d) * B_SZ + b] = vv;
            out[((size_t)b * J_UN + j) * D_SZ + d] = vv;   // [B,J,D,1]
        }
    }
}

__global__ void init_kernel(float* b_ij, float* agree) {
    int idx = blockIdx.x * 256 + threadIdx.x;
    if (idx < C_CH * J_UN) { b_ij[idx] = 0.f; agree[idx] = 0.f; }
}

extern "C" void kernel_launch(void* const* d_in, const int* in_sizes, int n_in,
                              void* d_out, int out_size, void* d_ws, size_t ws_size,
                              hipStream_t stream) {
    const float* x = (const float*)d_in[0];   // [128,32,4096]
    const float* W = (const float*)d_in[1];   // [4096,10,16,32]
    float* out = (float*)d_out;               // [128,10,16,1]
    float* ws  = (float*)d_ws;
    float* b_ij  = ws;                // 40960 f32
    float* agree = ws + 40960;        // 40960 f32
    float* c     = ws + 81920;        // 40960 f32
    float* s     = ws + 122880;       // 20480 f32
    float* v     = ws + 143360;       // 20480 f32

    size_t lds_bytes = (size_t)(CH_PER_WG * XS_CH_STRIDE) * 4 + (size_t)JD * B_SZ * 4;

    init_kernel<<<(C_CH * J_UN + 255) / 256, 256, 0, stream>>>(b_ij, agree);
    for (int t = 0; t < 3; ++t) {
        compute_c_kernel<<<1, 320, 0, stream>>>(b_ij, agree, c, s);
        pass_s_kernel<<<C_CH / CH_PER_WG, THREADS, lds_bytes, stream>>>(x, W, c, s);
        squash_kernel<<<1, 256, 0, stream>>>(s, v, out);
        if (t < 2)
            pass_agree_kernel<<<C_CH / CH_PER_WG, THREADS, lds_bytes, stream>>>(x, W, v, agree);
    }
}
